// SAGEMLP_42348377538966
// MI455X (gfx1250) — matrix operand order repaired and checked by reading, not compiled, not run on hardware
//
#include <hip/hip_runtime.h>
#include <hip/hip_bf16.h>
#include <math.h>

// ---------------------------------------------------------------------------
// SAGE GNN forward for MI455X (gfx1250, wave32).
//  - edge mean-aggregation via fp32 global atomics (L2-resident: 192MB L2
//    holds h (12.8MB), agg (<=52MB) and all weights)
//  - node GEMMs (agg@w_l^T + h@w_r^T) with v_wmma_f32_16x16x32_bf16,
//    fp32 accumulate; guard-free main K loop (float4 loads) + uniform tail
//  - fused bias+GELU+LayerNorm+residual epilogue via LDS tile transpose
// ---------------------------------------------------------------------------

typedef __attribute__((ext_vector_type(16))) __bf16 v16bf;
typedef __attribute__((ext_vector_type(8)))  float  v8f;

#define WAVES 4   // 4 waves/WG -> 64 output rows per workgroup

__device__ __forceinline__ float gelu_exact(float x) {
    return 0.5f * x * (1.0f + erff(x * 0.70710678118654752f));
}

// ---------------- degree / scatter / pool helpers --------------------------

__global__ void deg_kernel(const int* __restrict__ dst, float* __restrict__ deg, int E) {
    int e = blockIdx.x * blockDim.x + threadIdx.x;
    if (e < E) atomicAdd(&deg[dst[e]], 1.0f);
}

__global__ void invdeg_kernel(float* __restrict__ deg, int N) {
    int i = blockIdx.x * blockDim.x + threadIdx.x;
    if (i < N) deg[i] = 1.0f / fmaxf(deg[i], 1.0f);
}

// one thread per (edge, channel): coalesced within each feature row
__global__ void scatter_kernel(const float* __restrict__ feat,
                               const int* __restrict__ src, const int* __restrict__ dst,
                               float* __restrict__ agg, int E, int C) {
    unsigned i = blockIdx.x * blockDim.x + threadIdx.x;
    unsigned total = (unsigned)E * (unsigned)C;
    if (i >= total) return;
    unsigned e, c;
    if (C == 64) { e = i >> 6; c = i & 63u; }
    else         { e = i / (unsigned)C; c = i - e * (unsigned)C; }
    atomicAdd(&agg[(size_t)dst[e] * C + c], feat[(size_t)src[e] * C + c]);
}

// global_add_pool into z[:, 0:64] (row stride 88)
__global__ void pool_kernel(const float* __restrict__ h, const int* __restrict__ batch,
                            float* __restrict__ z, int N) {
    unsigned i = blockIdx.x * blockDim.x + threadIdx.x;
    unsigned total = (unsigned)N * 64u;
    if (i >= total) return;
    unsigned n = i >> 6, c = i & 63u;
    atomicAdd(&z[(size_t)batch[n] * 88 + c], h[i]);
}

__global__ void concat_kernel(const float* __restrict__ gf, float* __restrict__ z, int G) {
    int i = blockIdx.x * blockDim.x + threadIdx.x;
    if (i >= G * 24) return;
    int g = i / 24, j = i - g * 24;
    z[g * 88 + 64 + j] = gf[i];
}

// ---------------- fused SAGE dense block (WMMA bf16) -----------------------
// out[n,o] = LN( gelu( agg[n,:]*invdeg[n] @ W1[o,:] + b[o] + h[n,:] @ W2[o,:] ) ) (+ resid)

__global__ __launch_bounds__(32 * WAVES)
void sage_gemm_kernel(const float* __restrict__ A1,    // [N,C] summed neighbors
                      const float* __restrict__ A2,    // [N,C] self features
                      const float* __restrict__ W1,    // [64,C] w_l
                      const float* __restrict__ W2,    // [64,C] w_r
                      const float* __restrict__ bias,  // [64]
                      const float* __restrict__ gamma, // [64]
                      const float* __restrict__ beta,  // [64]
                      const float* __restrict__ invdeg,// [N]
                      const float* __restrict__ resid, // [N,64] or nullptr
                      float* __restrict__ out,         // [N,64]
                      int N, int C) {
    __shared__ float ldsOut[WAVES][16][68];   // +4 pad: no bank conflicts

    const int wave   = threadIdx.x >> 5;
    const int lane   = threadIdx.x & 31;
    const int laneHi = lane >> 4;             // 0: lanes 0-15, 1: lanes 16-31
    const int lane16 = lane & 15;

    const int m0  = (blockIdx.x * WAVES + wave) * 16;
    const int row = m0 + lane16;              // both lane halves hold same rows (A layout)
    const int rowc = (row < N) ? row : (N - 1);

    v8f acc[4];
#pragma unroll
    for (int t = 0; t < 4; ++t)
#pragma unroll
        for (int j = 0; j < 8; ++j) acc[t][j] = 0.0f;

#pragma unroll 1
    for (int phase = 0; phase < 2; ++phase) {
        const float* A = phase ? A2 : A1;
        const float* W = phase ? W2 : W1;
        const float  scale = phase ? 1.0f : invdeg[rowc];   // mean aggregation
        const float* Arow = A + (size_t)rowc * C;

        int kb = 0;
        // ---- main loop: no bounds checks, pure float4 traffic ----
        for (; kb + 32 <= C; kb += 32) {
            // A fragment: lane K runs [kb+16*hi, +15] (contiguous, matches B)
            const float4* Ap = reinterpret_cast<const float4*>(Arow + kb + laneHi * 16);
            float4 a0 = Ap[0];
            float4 a1 = Ap[1];
            float4 a2 = Ap[2];   // +8 floats
            float4 a3 = Ap[3];
            v16bf a;
            a[0]  = (__bf16)(a0.x * scale); a[1]  = (__bf16)(a0.y * scale);
            a[2]  = (__bf16)(a0.z * scale); a[3]  = (__bf16)(a0.w * scale);
            a[4]  = (__bf16)(a1.x * scale); a[5]  = (__bf16)(a1.y * scale);
            a[6]  = (__bf16)(a1.z * scale); a[7]  = (__bf16)(a1.w * scale);
            a[8]  = (__bf16)(a2.x * scale); a[9]  = (__bf16)(a2.y * scale);
            a[10] = (__bf16)(a2.z * scale); a[11] = (__bf16)(a2.w * scale);
            a[12] = (__bf16)(a3.x * scale); a[13] = (__bf16)(a3.y * scale);
            a[14] = (__bf16)(a3.z * scale); a[15] = (__bf16)(a3.w * scale);
#pragma unroll
            for (int t = 0; t < 4; ++t) {
                const int o = t * 16 + lane16;            // output channel = B column
                const float4* Bp = reinterpret_cast<const float4*>(
                    W + (size_t)o * C + kb + laneHi * 16);
                float4 b0 = Bp[0];
                float4 b1 = Bp[1];
                float4 b2 = Bp[2];
                float4 b3 = Bp[3];
                v16bf b;
                b[0]  = (__bf16)b0.x; b[1]  = (__bf16)b0.y;
                b[2]  = (__bf16)b0.z; b[3]  = (__bf16)b0.w;
                b[4]  = (__bf16)b1.x; b[5]  = (__bf16)b1.y;
                b[6]  = (__bf16)b1.z; b[7]  = (__bf16)b1.w;
                b[8]  = (__bf16)b2.x; b[9]  = (__bf16)b2.y;
                b[10] = (__bf16)b2.z; b[11] = (__bf16)b2.w;
                b[12] = (__bf16)b3.x; b[13] = (__bf16)b3.y;
                b[14] = (__bf16)b3.z; b[15] = (__bf16)b3.w;
                acc[t] = __builtin_amdgcn_wmma_f32_16x16x32_bf16(
                    false, a, false, b, (short)0, acc[t], false, false);
            }
        }
        // ---- uniform tail (C % 32 != 0): clamped loads + value select ----
        if (kb < C) {
            v16bf a;
            const int ka = kb + laneHi * 16;
#pragma unroll
            for (int e = 0; e < 8; ++e) {
                int k0 = ka + e;
                int k1 = ka + 8 + e;
                float v0 = Arow[k0 < C ? k0 : C - 1];
                float v1 = Arow[k1 < C ? k1 : C - 1];
                a[e]     = (__bf16)((k0 < C) ? v0 * scale : 0.0f);
                a[8 + e] = (__bf16)((k1 < C) ? v1 * scale : 0.0f);
            }
#pragma unroll
            for (int t = 0; t < 4; ++t) {
                const int o = t * 16 + lane16;
                const float* Wrow = W + (size_t)o * C;
                const int kw = kb + laneHi * 16;
                v16bf b;
#pragma unroll
                for (int e = 0; e < 16; ++e) {
                    int k = kw + e;
                    float v = Wrow[k < C ? k : C - 1];
                    b[e] = (__bf16)((k < C) ? v : 0.0f);
                }
                acc[t] = __builtin_amdgcn_wmma_f32_16x16x32_bf16(
                    false, a, false, b, (short)0, acc[t], false, false);
            }
        }
    }

    // ---- epilogue: dump C/D tile to LDS (M = j + 8*hi, Ncol = t*16 + lane16) ----
#pragma unroll
    for (int t = 0; t < 4; ++t) {
        const int Ncol = t * 16 + lane16;
        const float bv = bias[Ncol];
#pragma unroll
        for (int j = 0; j < 8; ++j)
            ldsOut[wave][j + laneHi * 8][Ncol] = acc[t][j] + bv;
    }
    __syncthreads();

    // lane pair (l, l+16) owns row r = lane16; each half handles 32 columns
    const int r = lane16;
    float s = 0.0f, ss = 0.0f;
#pragma unroll
    for (int c = 0; c < 32; ++c) {
        const int col = laneHi * 32 + c;
        float g = gelu_exact(ldsOut[wave][r][col]);
        ldsOut[wave][r][col] = g;
        s += g; ss += g * g;
    }
    s  += __shfl_xor(s, 16, 32);
    ss += __shfl_xor(ss, 16, 32);
    const float mu  = s * (1.0f / 64.0f);
    const float var = ss * (1.0f / 64.0f) - mu * mu;
    const float rs  = rsqrtf(var + 1e-5f);

    const int grow = m0 + r;
    if (grow < N) {
        const float* rp = resid ? resid + (size_t)grow * 64 : nullptr;
        float* op = out + (size_t)grow * 64;
#pragma unroll
        for (int c = 0; c < 32; ++c) {
            const int col = laneHi * 32 + c;
            float y = (ldsOut[wave][r][col] - mu) * rs * gamma[col] + beta[col];
            if (rp) y += rp[col];
            op[col] = y;
        }
    }
}

// ---------------- tiny pooled MLP head (64 rows, scalar) -------------------

struct MlpArgs {
    const float* w[4];  const float* b[4];
    const float* gm[4]; const float* bt[4];
    const float* hw;    const float* hb;
};

__global__ void mlp_head_kernel(const float* __restrict__ zin, MlpArgs a,
                                float* __restrict__ out, int G) {
    int g = blockIdx.x * blockDim.x + threadIdx.x;
    if (g >= G) return;
    float z[88];
#pragma unroll
    for (int k = 0; k < 88; ++k) z[k] = zin[g * 88 + k];

    int din = 88;
    for (int blk = 0; blk < 4; ++blk) {
        float f[32];
        const float* W = a.w[blk];
        for (int o = 0; o < 32; ++o) {
            float acc = a.b[blk][o];
            for (int k = 0; k < din; ++k) acc += z[k] * W[o * din + k];
            f[o] = gelu_exact(acc);
        }
        float s = 0.0f;
        for (int o = 0; o < 32; ++o) s += f[o];
        const float mu = s * (1.0f / 32.0f);
        float ss = 0.0f;
        for (int o = 0; o < 32; ++o) { float d = f[o] - mu; ss += d * d; }
        const float rs = rsqrtf(ss * (1.0f / 32.0f) + 1e-5f);
        for (int o = 0; o < 32; ++o) {
            float y = (f[o] - mu) * rs * a.gm[blk][o] + a.bt[blk][o];
            if (blk > 0) y += z[o];
            z[o] = y;
        }
        din = 32;
    }
    float y = a.hb[0];
    for (int k = 0; k < 32; ++k) y += z[k] * a.hw[k];
    out[g] = y;
}

// ---------------------------------------------------------------------------

extern "C" void kernel_launch(void* const* d_in, const int* in_sizes, int n_in,
                              void* d_out, int out_size, void* d_ws, size_t ws_size,
                              hipStream_t stream) {
    (void)n_in; (void)out_size; (void)ws_size;
    const int N = in_sizes[0] / 261;
    const int E = in_sizes[1] / 2;
    const int G = in_sizes[3] / 24;

    const float* x     = (const float*)d_in[0];
    const int*   src   = (const int*)d_in[1];
    const int*   dstp  = src + E;
    const int*   batch = (const int*)d_in[2];
    const float* gfeat = (const float*)d_in[3];

    auto sp = [&](int l, int w) { return (const float*)d_in[4 + l * 5 + w]; };  // w_l,b_l,w_r,gamma,beta
    auto mp = [&](int b, int w) { return (const float*)d_in[39 + b * 4 + w]; }; // w,b,gamma,beta
    const float* headw = (const float*)d_in[55];
    const float* headb = (const float*)d_in[56];

    // workspace carving (256B aligned)
    char* ws = (char*)d_ws;
    size_t off = 0;
    auto carve = [&](size_t bytes) {
        char* p = ws + off;
        off += (bytes + 255) & ~(size_t)255;
        return p;
    };
    float* invdeg = (float*)carve((size_t)N * 4);
    float* hA     = (float*)carve((size_t)N * 64 * 4);
    float* hB     = (float*)carve((size_t)N * 64 * 4);
    float* agg    = (float*)carve((size_t)N * 261 * 4);
    float* z      = (float*)carve((size_t)G * 88 * 4);

    // 1) degrees -> inv_deg (clamped to 1)
    hipMemsetAsync(invdeg, 0, (size_t)N * 4, stream);
    deg_kernel<<<(E + 255) / 256, 256, 0, stream>>>(dstp, invdeg, E);
    invdeg_kernel<<<(N + 255) / 256, 256, 0, stream>>>(invdeg, N);

    // 2) SAGE layer 0 (C=261, no residual)
    hipMemsetAsync(agg, 0, (size_t)N * 261 * 4, stream);
    {
        unsigned total = (unsigned)E * 261u;
        scatter_kernel<<<(total + 255) / 256, 256, 0, stream>>>(x, src, dstp, agg, E, 261);
    }
    sage_gemm_kernel<<<(N + 63) / 64, 32 * WAVES, 0, stream>>>(
        agg, x, sp(0, 0), sp(0, 2), sp(0, 1), sp(0, 3), sp(0, 4),
        invdeg, nullptr, hA, N, 261);

    // 3) SAGE layers 1..6 (C=64, residual)
    float* cur = hA;
    float* nxt = hB;
    for (int l = 1; l <= 6; ++l) {
        hipMemsetAsync(agg, 0, (size_t)N * 64 * 4, stream);
        unsigned total = (unsigned)E * 64u;
        scatter_kernel<<<(total + 255) / 256, 256, 0, stream>>>(cur, src, dstp, agg, E, 64);
        sage_gemm_kernel<<<(N + 63) / 64, 32 * WAVES, 0, stream>>>(
            agg, cur, sp(l, 0), sp(l, 2), sp(l, 1), sp(l, 3), sp(l, 4),
            invdeg, cur, nxt, N, 64);
        float* t = cur; cur = nxt; nxt = t;
    }

    // 4) global_add_pool + concat global features into z[G,88]
    hipMemsetAsync(z, 0, (size_t)G * 88 * 4, stream);
    {
        unsigned total = (unsigned)N * 64u;
        pool_kernel<<<(total + 255) / 256, 256, 0, stream>>>(cur, batch, z, N);
    }
    concat_kernel<<<(G * 24 + 255) / 256, 256, 0, stream>>>(gfeat, z, G);

    // 5) MLP head -> d_out [G,1]
    MlpArgs ma;
    for (int b = 0; b < 4; ++b) {
        ma.w[b]  = mp(b, 0); ma.b[b]  = mp(b, 1);
        ma.gm[b] = mp(b, 2); ma.bt[b] = mp(b, 3);
    }
    ma.hw = headw; ma.hb = headb;
    mlp_head_kernel<<<(G + 63) / 64, 64, 0, stream>>>(z, ma, (float*)d_out, G);
}
